// AttentionLayer_60172491817534
// MI455X (gfx1250) — compile-verified, matrix operand order checked by reading, not run on hardware
//
#include <hip/hip_runtime.h>
#include <hip/hip_bf16.h>

#define BB 2
#define TT 2048
#define CC 2048
#define HH 16
#define HD 128
#define MM (BB*TT)   // 4096 rows for the two big GEMMs
#define C3 (3*CC)    // 6144

typedef __bf16 bf16_t;
typedef __attribute__((ext_vector_type(16))) __bf16 v16bf;
typedef __attribute__((ext_vector_type(8)))  __bf16 bf16x8;
typedef __attribute__((ext_vector_type(8)))  float  v8f;

// Load one 16x32 bf16 WMMA operand fragment (A-layout; B-layout is the mirror
// with N fixed per lane) from a row-major [rows x ld] matrix where the
// per-lane-fixed index is the memory row and K runs along the row.
// Per-lane data = two contiguous 8-element (16B) chunks:
//   lanes 0-15 : K = k0+[0..7]  and k0+[16..23]
//   lanes 16-31: K = k0+[8..15] and k0+[24..31]
__device__ __forceinline__ v16bf load_frag_rowK(const bf16_t* base, int ld,
                                                int row0, int k0, int lane) {
  const int r  = row0 + (lane & 15);
  const int ks = k0 + ((lane >> 4) << 3);
  const bf16_t* p = base + (size_t)r * ld + ks;
  bf16x8 lo = *(const bf16x8*)(p);
  bf16x8 hi = *(const bf16x8*)(p + 16);
  v16bf f;
#pragma unroll
  for (int i = 0; i < 8; ++i) { f[i] = lo[i]; f[i + 8] = hi[i]; }
  return f;
}

__device__ __forceinline__ v8f wmma_bf16(v16bf a, v16bf b, v8f c) {
  return __builtin_amdgcn_wmma_f32_16x16x32_bf16(false, a, false, b,
                                                 (short)0, c, false, false);
}

// ---------------------------------------------------------------- convert --
__global__ void f32_to_bf16(const float* __restrict__ in,
                            bf16_t* __restrict__ out, int n) {
  int i = blockIdx.x * blockDim.x + threadIdx.x;
  if (i < n) out[i] = (bf16_t)in[i];
}

// ------------------------------------------------------------------- GEMM --
// out[m,n] = sum_k A[m,k] * W[n,k]  (+ bias[n])
// A: [M,K] bf16 row-major, W: [N,K] bf16 row-major (torch Linear layout).
// One wave computes a 64x64 tile (4x4 accumulator sub-tiles).
// B fragments are streamed (one live at a time) to keep VGPR pressure low:
// live state ~= 128 acc + 32 A-frag + 8 B-frag + addressing < 200 VGPRs.
template <bool BF16_OUT>
__global__ __launch_bounds__(128) void gemm_xWT(const bf16_t* __restrict__ A,
                                                const bf16_t* __restrict__ W,
                                                const float* __restrict__ bias,
                                                void* __restrict__ out,
                                                int M, int N, int K) {
  const int lane = threadIdx.x & 31;
  const int wid  = blockIdx.x * (blockDim.x >> 5) + (threadIdx.x >> 5);
  const int mt   = M >> 6;
  const int m0   = (wid % mt) * 64;
  const int n0   = (wid / mt) * 64;
  if (n0 >= N) return;

  // hoisted prefetch base rows (add-only addressing in the loop)
  const bf16_t* pfA = A + (size_t)(m0 + lane) * K;
  const bf16_t* pfW = W + (size_t)(n0 + lane) * K;

  v8f acc[4][4] = {};
  for (int k0 = 0; k0 < K; k0 += 32) {
    if (k0 + 128 < K) {  // stream prefetch (-> global_prefetch_b8)
      __builtin_prefetch(pfA + k0 + 128, 0, 1);
      __builtin_prefetch(pfW + k0 + 128, 0, 1);
    }
    v16bf a[4];
#pragma unroll
    for (int i = 0; i < 4; ++i) a[i] = load_frag_rowK(A, K, m0 + 16 * i, k0, lane);
#pragma unroll
    for (int j = 0; j < 4; ++j) {
      v16bf b = load_frag_rowK(W, K, n0 + 16 * j, k0, lane);
#pragma unroll
      for (int i = 0; i < 4; ++i)
        acc[i][j] = wmma_bf16(a[i], b, acc[i][j]);
    }
  }
  // C/D layout: VGPR r, lane L -> (M = r + 8*(L>=16), N = L&15)
  const int nlo = lane & 15;
  const int rhi = (lane >> 4) * 8;
#pragma unroll
  for (int i = 0; i < 4; ++i)
#pragma unroll
    for (int j = 0; j < 4; ++j) {
      int n = n0 + 16 * j + nlo;
      float bv = bias ? bias[n] : 0.0f;
#pragma unroll
      for (int r = 0; r < 8; ++r) {
        int m = m0 + 16 * i + r + rhi;
        float v = acc[i][j][r] + bv;
        if (BF16_OUT) ((bf16_t*)out)[(size_t)m * N + n] = (bf16_t)v;
        else          ((float*)out)[(size_t)m * N + n] = v;
      }
    }
}

// ----------------------------------------------------------- V transpose --
// vT[b][h][d][t] = qkv[b][t][2C + h*HD + d]
__global__ void transpose_v(const bf16_t* __restrict__ qkv,
                            bf16_t* __restrict__ vT) {
  size_t i = (size_t)blockIdx.x * blockDim.x + threadIdx.x;
  int d = (int)(i % HD);
  int h = (int)((i / HD) % HH);
  int t = (int)((i / CC) % TT);
  int b = (int)(i / ((size_t)CC * TT));
  bf16_t v = qkv[((size_t)b * TT + t) * C3 + 2 * CC + h * HD + d];
  vT[(((size_t)(b * HH + h)) * HD + d) * TT + t] = v;
}

// ------------------------------------------------------- flash attention --
// One wave (single-wave workgroup) owns 16 query rows of one (b,h).
__global__ __launch_bounds__(32) void attn_kernel(const bf16_t* __restrict__ qkv,
                                                  const bf16_t* __restrict__ vT,
                                                  bf16_t* __restrict__ y) {
  __shared__ __attribute__((aligned(16))) bf16_t lp[16 * 32];
  const int lane = threadIdx.x & 31;
  const int QT = TT / 16;
  const int qt = blockIdx.x % QT;
  const int h  = (blockIdx.x / QT) % HH;
  const int b  = blockIdx.x / (QT * HH);
  const int q0 = qt * 16;

  const bf16_t* qbase = qkv + (size_t)b * TT * C3 + h * HD;       // rows t, ld=3C
  const bf16_t* kbase = qbase + CC;
  const bf16_t* vtb   = vT + ((size_t)(b * HH + h)) * HD * TT;    // rows d, ld=T

  v16bf qf[4];
#pragma unroll
  for (int kb = 0; kb < 4; ++kb)
    qf[kb] = load_frag_rowK(qbase, C3, q0, kb * 32, lane);

  v8f o[8] = {};
  float mrow[8], lrow[8];
#pragma unroll
  for (int r = 0; r < 8; ++r) { mrow[r] = -1e30f; lrow[r] = 0.0f; }

  const float scale = 0.08838834764831845f;  // 1/sqrt(128)
  const int nlo = lane & 15;
  const int rhi = (lane >> 4) * 8;

  for (int j0 = 0; j0 < q0 + 16; j0 += 32) {
    // ---- S = Q K^T (16x32), K-dim = HD = 128 -> 8 WMMAs
    v8f s[2] = {};
#pragma unroll
    for (int nt = 0; nt < 2; ++nt)
#pragma unroll
      for (int kb = 0; kb < 4; ++kb) {
        v16bf kf = load_frag_rowK(kbase, C3, j0 + 16 * nt, kb * 32, lane);
        s[nt] = wmma_bf16(qf[kb], kf, s[nt]);
      }
    // ---- scale + causal mask
#pragma unroll
    for (int nt = 0; nt < 2; ++nt)
#pragma unroll
      for (int r = 0; r < 8; ++r) {
        int kc = j0 + 16 * nt + nlo;
        int qr = q0 + r + rhi;
        float sv = s[nt][r] * scale;
        s[nt][r] = (kc <= qr) ? sv : -1e30f;
      }
    // ---- online softmax (row stats live per lane-half; xor 1,2,4,8 stays in half)
    float alpha[8];
#pragma unroll
    for (int r = 0; r < 8; ++r) {
      float v = fmaxf(s[0][r], s[1][r]);
#pragma unroll
      for (int off = 1; off < 16; off <<= 1) v = fmaxf(v, __shfl_xor(v, off, 32));
      float mn = fmaxf(mrow[r], v);
      alpha[r] = __expf(mrow[r] - mn);
      mrow[r] = mn;
    }
#pragma unroll
    for (int nt = 0; nt < 2; ++nt)
#pragma unroll
      for (int r = 0; r < 8; ++r)
        s[nt][r] = __expf(s[nt][r] - mrow[r]);
#pragma unroll
    for (int r = 0; r < 8; ++r) {
      float rs = s[0][r] + s[1][r];
#pragma unroll
      for (int off = 1; off < 16; off <<= 1) rs += __shfl_xor(rs, off, 32);
      lrow[r] = lrow[r] * alpha[r] + rs;
    }
#pragma unroll
    for (int dt = 0; dt < 8; ++dt)
#pragma unroll
      for (int r = 0; r < 8; ++r) o[dt][r] *= alpha[r];

    // ---- P: D-layout -> A-layout via LDS (row-major [16][32] bf16)
    __syncthreads();
#pragma unroll
    for (int nt = 0; nt < 2; ++nt)
#pragma unroll
      for (int r = 0; r < 8; ++r)
        lp[(r + rhi) * 32 + nt * 16 + nlo] = (bf16_t)s[nt][r];
    __syncthreads();
    v16bf pf = load_frag_rowK(lp, 32, 0, 0, lane);

    // ---- O += P V : 8 output d-tiles, K=32 -> 8 WMMAs
#pragma unroll
    for (int dt = 0; dt < 8; ++dt) {
      v16bf vf = load_frag_rowK(vtb, TT, dt * 16, j0, lane);
      o[dt] = wmma_bf16(pf, vf, o[dt]);
    }
  }
  // ---- normalize and write y[b][t][h*HD + d]
  bf16_t* yb = y + (size_t)b * TT * CC + h * HD;
#pragma unroll
  for (int r = 0; r < 8; ++r) {
    float inv = 1.0f / lrow[r];
    int t = q0 + r + rhi;
#pragma unroll
    for (int dt = 0; dt < 8; ++dt)
      yb[(size_t)t * CC + dt * 16 + nlo] = (bf16_t)(o[dt][r] * inv);
  }
}

// ---------------------------------------------------------------- launch --
extern "C" void kernel_launch(void* const* d_in, const int* in_sizes, int n_in,
                              void* d_out, int out_size, void* d_ws, size_t ws_size,
                              hipStream_t stream) {
  const float* x      = (const float*)d_in[0];
  const float* w_attn = (const float*)d_in[1];
  const float* w_proj = (const float*)d_in[2];
  const float* b_proj = (const float*)d_in[3];
  float* out = (float*)d_out;

  bf16_t* xb   = (bf16_t*)d_ws;                 // [MM, CC]
  bf16_t* wab  = xb   + (size_t)MM * CC;        // [C3, CC]
  bf16_t* wpb  = wab  + (size_t)C3 * CC;        // [CC, CC]
  bf16_t* qkvb = wpb  + (size_t)CC * CC;        // [MM, C3]
  bf16_t* vTb  = qkvb + (size_t)MM * C3;        // [B,H,HD,TT]
  bf16_t* yb   = vTb  + (size_t)MM * CC;        // [MM, CC]

  int n1 = MM * CC;   // 8.4M
  int n2 = C3 * CC;   // 12.6M
  int n3 = CC * CC;   // 4.2M
  f32_to_bf16<<<(n1 + 255) / 256, 256, 0, stream>>>(x, xb, n1);
  f32_to_bf16<<<(n2 + 255) / 256, 256, 0, stream>>>(w_attn, wab, n2);
  f32_to_bf16<<<(n3 + 255) / 256, 256, 0, stream>>>(w_proj, wpb, n3);

  {  // qkv = x @ w_attn^T  (M=4096, N=6144, K=2048)
    int waves = (MM / 64) * (C3 / 64);   // 6144 waves
    gemm_xWT<true><<<waves / 4, 128, 0, stream>>>(xb, wab, nullptr, qkvb,
                                                  MM, C3, CC);
  }
  transpose_v<<<(n1 + 255) / 256, 256, 0, stream>>>(qkvb, vTb);
  attn_kernel<<<BB * HH * (TT / 16), 32, 0, stream>>>(qkvb, vTb, yb);
  {  // out = y @ w_proj^T + b_proj  (M=4096, N=2048, K=2048), fp32 out
    int waves = (MM / 64) * (CC / 64);   // 2048 waves
    gemm_xWT<false><<<waves / 4, 128, 0, stream>>>(yb, wpb, b_proj, out,
                                                   MM, CC, CC);
  }
}
